// TransConvLayer_11974368821796
// MI455X (gfx1250) — matrix operand order repaired and checked by reading, not compiled, not run on hardware
//
#include <hip/hip_runtime.h>
#include <hip/hip_bf16.h>

typedef __attribute__((ext_vector_type(16))) _Float16 v16h;
typedef __attribute__((ext_vector_type(8)))  _Float16 v8h;
typedef __attribute__((ext_vector_type(8)))  float    v8f;
typedef __attribute__((ext_vector_type(4)))  int      v4i;

#define NB    16
#define LSEQ  4096
#define IN_C  256
#define OUT_C 128
#define NH    8
#define DH    1024          // OUT_C * NH
#define TOTAL 65536         // NB * LSEQ
#define KC    32            // WMMA K per instruction (f16)
#define KS    64            // K per LDS stage in qkv_gemm
#define MT    128
#define NT    128
#define KSPLIT 8            // split-K factor for kv_gemm

// ---------------- CDNA5 async global->LDS copy (ASYNCcnt path) ----------------
#if defined(__has_builtin)
#  if __has_builtin(__builtin_amdgcn_global_load_async_to_lds_b128)
#    define HAVE_ASYNC_LDS 1
#  endif
#endif
#ifndef HAVE_ASYNC_LDS
#  define HAVE_ASYNC_LDS 0
#endif

__device__ __forceinline__ void cp16(void* dst_lds, const void* src_global) {
#if HAVE_ASYNC_LDS
    __builtin_amdgcn_global_load_async_to_lds_b128(
        (__attribute__((address_space(1))) v4i*)src_global,
        (__attribute__((address_space(3))) v4i*)dst_lds, 0, 0);
#else
    *(v8h*)dst_lds = *(const v8h*)src_global;
#endif
}
__device__ __forceinline__ void cp16_wait() {
#if HAVE_ASYNC_LDS
#  if __has_builtin(__builtin_amdgcn_s_wait_asynccnt)
    __builtin_amdgcn_s_wait_asynccnt(0);
#  else
    asm volatile("s_wait_asynccnt 0" ::: "memory");
#  endif
#endif
}

// ---- fragment loaders (CDNA5 V_WMMA_F32_16X16X32_F16 layouts, wave32) ----
// A (16x32 MxK): lane m=lane&15, hh=lane>>4; halves 0..7 = K hh*8.., 8..15 = K 16+hh*8..
__device__ __forceinline__ v16h load_a_frag(const _Float16* s, int row0, int kb, int ld, int lane) {
    int ln = lane & 15, hh = lane >> 4;
    const _Float16* p = s + (row0 + ln) * ld + kb;
    union { v16h v; v8h h[2]; } u;
    u.h[0] = *(const v8h*)(p + hh * 8);
    u.h[1] = *(const v8h*)(p + 16 + hh * 8);
    return u.v;
}
// B (32x16 KxN) staged transposed as Bt[N][K]: lane n; halves 0..7 = K hh*8.., 8..15 = K 16+hh*8..
__device__ __forceinline__ v16h load_b_frag(const _Float16* s, int row0, int kb, int ld, int lane) {
    int ln = lane & 15, hh = lane >> 4;
    const _Float16* p = s + (row0 + ln) * ld + kb;
    union { v16h v; v8h h[2]; } u;
    u.h[0] = *(const v8h*)(p + hh * 8);
    u.h[1] = *(const v8h*)(p + 16 + hh * 8);
    return u.v;
}
__device__ __forceinline__ v8f wmma_f16(v16h a, v16h b, v8f c) {
    return __builtin_amdgcn_wmma_f32_16x16x32_f16(false, a, false, b, (short)0, c, false, false);
}

// ---- kernel 0: transpose+convert weights: Wt[mat][n][k] f16 ----
__global__ __launch_bounds__(256) void wt_kernel(const float* __restrict__ Wq,
                                                 const float* __restrict__ Wk,
                                                 const float* __restrict__ Wv,
                                                 _Float16* __restrict__ Wt) {
    int g = blockIdx.x * 256 + threadIdx.x;            // 3*1024*256
    int mat = g / (DH * IN_C);
    int e   = g % (DH * IN_C);
    int n = e / IN_C, k = e % IN_C;
    const float* W = (mat == 0) ? Wq : (mat == 1) ? Wk : Wv;
    Wt[g] = (_Float16)W[k * DH + n];
}

// ---- kernel 1: fused QKV GEMM (X f32 -> Y f16) + Q/K sumsq + K colsum ----
// 128x128 tile, 8 waves (2x4), KS=64 stage, reg-prefetched A, async-copied B.
__global__ __launch_bounds__(256) void qkv_gemm(const float* __restrict__ X,
                                                const _Float16* __restrict__ WtAll,
                                                const float* __restrict__ bq,
                                                const float* __restrict__ bk,
                                                const float* __restrict__ bv,
                                                _Float16* __restrict__ Qh,
                                                _Float16* __restrict__ Kh,
                                                _Float16* __restrict__ Vh,
                                                float* __restrict__ pQ,
                                                float* __restrict__ pK,
                                                float* __restrict__ pcs) {
    const int mat   = blockIdx.z;
    const int mBase = blockIdx.y * MT;
    const int nBase = blockIdx.x * NT;
    const _Float16* Wt  = WtAll + (size_t)mat * DH * IN_C;
    const float* bias   = (mat == 0) ? bq : (mat == 1) ? bk : bv;
    _Float16* Y         = (mat == 0) ? Qh : (mat == 1) ? Kh : Vh;

    __shared__ __align__(64) _Float16 sA[MT * KS];     // 16 KB
    __shared__ __align__(64) _Float16 sB[NT * KS];     // 16 KB
    __shared__ float red[256];
    __shared__ float cs[NT * 4];

    const int tid = threadIdx.x, lane = tid & 31, wave = tid >> 5;
    const int waveM = (wave >> 2) * 64;
    const int waveN = (wave & 3) * 32;
    const int ln = lane & 15, hh = lane >> 4;

    v8f acc[4][2] = {};
    float4 ar[8];                                      // A prefetch regs (32 f32)

    auto loadA = [&](int kk) {
        #pragma unroll
        for (int i = 0; i < 4; i++) {                  // 1024 chunks of 8 halfs
            int c = tid * 4 + i; int r = c >> 3, col = (c & 7) * 8;
            const float* src = X + (size_t)(mBase + r) * IN_C + kk + col;
            ar[2*i]   = *(const float4*)src;
            ar[2*i+1] = *(const float4*)(src + 4);
        }
    };
    auto storeA = [&]() {
        #pragma unroll
        for (int i = 0; i < 4; i++) {
            int c = tid * 4 + i; int r = c >> 3, col = (c & 7) * 8;
            float4 f0 = ar[2*i], f1 = ar[2*i+1];
            v8h hv;
            hv[0]=(_Float16)f0.x; hv[1]=(_Float16)f0.y; hv[2]=(_Float16)f0.z; hv[3]=(_Float16)f0.w;
            hv[4]=(_Float16)f1.x; hv[5]=(_Float16)f1.y; hv[6]=(_Float16)f1.z; hv[7]=(_Float16)f1.w;
            *(v8h*)&sA[r * KS + col] = hv;
        }
    };
    auto copyB = [&](int kk) {
        #pragma unroll
        for (int i = 0; i < 4; i++) {
            int c = tid * 4 + i; int r = c >> 3, col = (c & 7) * 8;
            cp16(&sB[r * KS + col], Wt + (size_t)(nBase + r) * IN_C + kk + col);
        }
    };

    loadA(0);
    for (int s = 0; s < IN_C / KS; s++) {
        const int kk = s * KS;
        copyB(kk);                                     // sB free: last compute barrier
        storeA();
        if (s + 1 < IN_C / KS) loadA(kk + KS);         // overlaps with wait+compute
        cp16_wait();
        __syncthreads();
        #pragma unroll
        for (int kb = 0; kb < KS; kb += KC) {
            v16h a[4], b[2];
            #pragma unroll
            for (int i = 0; i < 4; i++) a[i] = load_a_frag(sA, waveM + i * 16, kb, KS, lane);
            #pragma unroll
            for (int j = 0; j < 2; j++) b[j] = load_b_frag(sB, waveN + j * 16, kb, KS, lane);
            #pragma unroll
            for (int i = 0; i < 4; i++)
                #pragma unroll
                for (int j = 0; j < 2; j++) acc[i][j] = wmma_f16(a[i], b[j], acc[i][j]);
        }
        __syncthreads();
    }

    float myss = 0.f;
    float mycs[2] = {0.f, 0.f};
    #pragma unroll
    for (int i = 0; i < 4; i++) {
        #pragma unroll
        for (int j = 0; j < 2; j++) {
            int colg = nBase + waveN + j * 16 + ln;
            float bv_ = bias[colg];
            #pragma unroll
            for (int v = 0; v < 8; v++) {
                int rowg = mBase + waveM + i * 16 + hh * 8 + v;
                float val = acc[i][j][v] + bv_;
                Y[(size_t)rowg * DH + colg] = (_Float16)val;
                myss    += val * val;
                mycs[j] += val;
            }
        }
    }
    if (mat < 2) {
        red[tid] = myss;
        __syncthreads();
        for (int s = 128; s > 0; s >>= 1) { if (tid < s) red[tid] += red[tid + s]; __syncthreads(); }
        if (tid == 0) ((mat == 0) ? pQ : pK)[blockIdx.y * gridDim.x + blockIdx.x] = red[0];
        if (mat == 1) {
            #pragma unroll
            for (int j = 0; j < 2; j++)
                cs[(waveN + j * 16 + ln) * 4 + (wave >> 2) * 2 + hh] = mycs[j];
            __syncthreads();
            if (tid < NT) {
                float s = cs[tid*4] + cs[tid*4+1] + cs[tid*4+2] + cs[tid*4+3];
                int bb = blockIdx.y >> 5, mt = blockIdx.y & 31;
                pcs[(size_t)((bb * 32 + mt) * DH) + nBase + tid] = s;
            }
        }
    }
}

// ---- kernel 2a: deterministic reduce of sumsq -> inv norms ----
__global__ __launch_bounds__(256) void reduce_norms(const float* __restrict__ pQ,
                                                    const float* __restrict__ pK,
                                                    float* __restrict__ scal) {
    __shared__ float red[256];
    int tid = threadIdx.x;
    float s = 0.f;
    for (int i = tid; i < 4096; i += 256) s += pQ[i];
    red[tid] = s; __syncthreads();
    for (int st = 128; st > 0; st >>= 1) { if (tid < st) red[tid] += red[tid + st]; __syncthreads(); }
    if (tid == 0) scal[0] = 1.0f / sqrtf(red[0]);
    __syncthreads();
    s = 0.f;
    for (int i = tid; i < 4096; i += 256) s += pK[i];
    red[tid] = s; __syncthreads();
    for (int st = 128; st > 0; st >>= 1) { if (tid < st) red[tid] += red[tid + st]; __syncthreads(); }
    if (tid == 0) scal[1] = 1.0f / sqrtf(red[0]);
}

// ---- kernel 2b: K column sums per batch ----
__global__ __launch_bounds__(256) void ksum_kernel(const float* __restrict__ pcs,
                                                   float* __restrict__ ksum) {
    int g = blockIdx.x * 256 + threadIdx.x;            // 16*1024
    int b = g >> 10, n = g & 1023;
    float s = 0.f;
    for (int j = 0; j < 32; j++) s += pcs[(size_t)((b * 32 + j) << 10) + n];
    ksum[g] = s;
}

// ---- kernel 2c: denominators den[t][h] = invQ*invK*<Q,Ksum> + L ----
__global__ __launch_bounds__(256) void den_kernel(const _Float16* __restrict__ Qh,
                                                  const float* __restrict__ ksum,
                                                  const float* __restrict__ scal,
                                                  float* __restrict__ den) {
    int g = blockIdx.x * 256 + threadIdx.x;            // TOTAL*NH
    int t = g >> 3, h = g & 7, b = t >> 12;
    const _Float16* q = Qh + (size_t)t * DH + h * OUT_C;
    const float* ks = ksum + b * DH + h * OUT_C;
    float s = 0.f;
    #pragma unroll
    for (int c = 0; c < OUT_C; c += 8) {
        v8h qv = *(const v8h*)(q + c);
        #pragma unroll
        for (int e = 0; e < 8; e++) s += (float)qv[e] * ks[c + e];
    }
    den[g] = scal[0] * scal[1] * s + (float)LSEQ;
}

// ---- kernel 3: split-K partial KV[b,h] = K^T V over an L/8 chunk (f32) ----
__global__ __launch_bounds__(256) void kv_gemm(const _Float16* __restrict__ Kh,
                                               const _Float16* __restrict__ Vh,
                                               float* __restrict__ pKV) {
    const int bh = blockIdx.x, split = blockIdx.y;
    const int b = bh >> 3, h = bh & 7;
    const int l0 = split * (LSEQ / KSPLIT);            // 512-chunk
    __shared__ __align__(64) _Float16 sA[OUT_C * KC];  // K^T tile [m][l]
    __shared__ __align__(64) _Float16 sB[OUT_C * KC];  // V^T tile [d][l]
    const int tid = threadIdx.x, lane = tid & 31, wave = tid >> 5;
    const int waveM = (wave >> 2) * 64, waveN = (wave & 3) * 32;
    const int ln = lane & 15, hh = lane >> 4;

    v8f acc[4][2] = {};
    for (int kk = l0; kk < l0 + LSEQ / KSPLIT; kk += KC) {
        #pragma unroll
        for (int i = 0; i < 2; i++) {                  // 512 chunks: (lc, m0)
            int c = tid * 2 + i;
            int lc = c >> 4, m0 = (c & 15) * 8;
            size_t src = (size_t)(b * LSEQ + kk + lc) * DH + h * OUT_C + m0;
            v8h kv = *(const v8h*)(Kh + src);
            v8h vv = *(const v8h*)(Vh + src);
            #pragma unroll
            for (int q = 0; q < 8; q++) {              // transpose into LDS
                sA[(m0 + q) * KC + lc] = kv[q];
                sB[(m0 + q) * KC + lc] = vv[q];
            }
        }
        __syncthreads();
        v16h a[4], bb[2];
        #pragma unroll
        for (int i = 0; i < 4; i++) a[i] = load_a_frag(sA, waveM + i * 16, 0, KC, lane);
        #pragma unroll
        for (int j = 0; j < 2; j++) bb[j] = load_b_frag(sB, waveN + j * 16, 0, KC, lane);
        #pragma unroll
        for (int i = 0; i < 4; i++)
            #pragma unroll
            for (int j = 0; j < 2; j++) acc[i][j] = wmma_f16(a[i], bb[j], acc[i][j]);
        __syncthreads();
    }
    // partial stored d-major so the reduce is fully contiguous: pKV[bh][split][d][m]
    float* dst = pKV + (((size_t)bh * KSPLIT + split) << 14);
    #pragma unroll
    for (int i = 0; i < 4; i++) {
        #pragma unroll
        for (int j = 0; j < 2; j++) {
            int d = waveN + j * 16 + ln;
            *(v8f*)(dst + (size_t)d * OUT_C + waveM + i * 16 + hh * 8) = acc[i][j];
        }
    }
}

// ---- kernel 3b: deterministic split-K reduce, scale by invK, cvt f16 ----
__global__ __launch_bounds__(256) void kv_reduce(const float* __restrict__ pKV,
                                                 const float* __restrict__ scal,
                                                 _Float16* __restrict__ KVt) {
    int g = blockIdx.x * 256 + threadIdx.x;            // NB*NH*128*128
    int bh = g >> 14, e = g & 16383;
    float s = 0.f;
    #pragma unroll
    for (int sp = 0; sp < KSPLIT; sp++)
        s += pKV[(((size_t)bh * KSPLIT + sp) << 14) + e];
    KVt[g] = (_Float16)(scal[1] * s);
}

// ---- kernel 4: per (b, mtile): loop h { Q@KV'; fuse +L*V, /den }, mean over h ----
__global__ __launch_bounds__(256) void attn_kernel(const _Float16* __restrict__ Qh,
                                                   const _Float16* __restrict__ Vh,
                                                   const _Float16* __restrict__ KVt,
                                                   const float* __restrict__ den,
                                                   const float* __restrict__ scal,
                                                   float* __restrict__ out) {
    const int b = blockIdx.y, mt = blockIdx.x;
    const int tBase = b * LSEQ + mt * MT;
    __shared__ __align__(64) _Float16 sQ[MT * OUT_C];   // 32 KB
    __shared__ __align__(64) _Float16 sKV[OUT_C * OUT_C]; // 32 KB (Bt layout [d][m])
    __shared__ __align__(64) _Float16 sV[MT * OUT_C];   // 32 KB
    __shared__ float sDen[MT];
    const int tid = threadIdx.x, lane = tid & 31, wave = tid >> 5;
    const int waveM = (wave >> 2) * 64, waveN = (wave & 3) * 32;
    const int ln = lane & 15, hh = lane >> 4;
    const float invQ = scal[0];

    v8f fac[4][2] = {};
    for (int h = 0; h < NH; h++) {
        const _Float16* KVb = KVt + (size_t)(b * NH + h) * OUT_C * OUT_C;
        #pragma unroll
        for (int i = 0; i < 8; i++) {                  // stage full tiles per head
            int c = i * 256 + tid;                     // 2048 chunks of 8 halfs
            int r = c >> 4, d0 = (c & 15) * 8;
            size_t o = (size_t)(tBase + r) * DH + h * OUT_C + d0;
            cp16(&sQ[r * OUT_C + d0],  Qh + o);
            cp16(&sV[r * OUT_C + d0],  Vh + o);
            cp16(&sKV[r * OUT_C + d0], KVb + (size_t)r * OUT_C + d0);
        }
        if (tid < MT) sDen[tid] = den[(size_t)(tBase + tid) * NH + h];
        v8f acc[4][2] = {};
        cp16_wait();
        __syncthreads();
        #pragma unroll
        for (int kb = 0; kb < OUT_C; kb += KC) {       // 4 k-steps, no barriers
            v16h a[4], bb[2];
            #pragma unroll
            for (int i = 0; i < 4; i++) a[i] = load_a_frag(sQ, waveM + i * 16, kb, OUT_C, lane);
            #pragma unroll
            for (int j = 0; j < 2; j++) bb[j] = load_b_frag(sKV, waveN + j * 16, kb, OUT_C, lane);
            #pragma unroll
            for (int i = 0; i < 4; i++)
                #pragma unroll
                for (int j = 0; j < 2; j++) acc[i][j] = wmma_f16(a[i], bb[j], acc[i][j]);
        }
        #pragma unroll
        for (int i = 0; i < 4; i++) {
            #pragma unroll
            for (int j = 0; j < 2; j++) {
                int d = waveN + j * 16 + ln;
                #pragma unroll
                for (int v = 0; v < 8; v++) {
                    int r = waveM + i * 16 + hh * 8 + v;
                    float num = invQ * acc[i][j][v]
                              + (float)LSEQ * (float)sV[r * OUT_C + d];
                    fac[i][j][v] += num / sDen[r];
                }
            }
        }
        __syncthreads();                               // before restaging LDS
    }
    #pragma unroll
    for (int i = 0; i < 4; i++) {
        #pragma unroll
        for (int j = 0; j < 2; j++) {
            int d = waveN + j * 16 + ln;
            #pragma unroll
            for (int v = 0; v < 8; v++) {
                int r = waveM + i * 16 + hh * 8 + v;
                out[(size_t)(tBase + r) * OUT_C + d] = fac[i][j][v] * 0.125f;
            }
        }
    }
}

extern "C" void kernel_launch(void* const* d_in, const int* in_sizes, int n_in,
                              void* d_out, int out_size, void* d_ws, size_t ws_size,
                              hipStream_t stream) {
    const float* x  = (const float*)d_in[0];
    // d_in[1] = batch indices: fixed repeat(arange(16), 4096) -> implicit in layout
    const float* Wq = (const float*)d_in[2];
    const float* bq = (const float*)d_in[3];
    const float* Wk = (const float*)d_in[4];
    const float* bk = (const float*)d_in[5];
    const float* Wv = (const float*)d_in[6];
    const float* bv = (const float*)d_in[7];
    float* out = (float*)d_out;

    char* wsb = (char*)d_ws;
    auto carve = [&](size_t bytes) {
        char* p = wsb; wsb += (bytes + 255) & ~(size_t)255; return p;
    };
    _Float16* Wt  = (_Float16*)carve((size_t)3 * DH * IN_C * 2);
    _Float16* Qh  = (_Float16*)carve((size_t)TOTAL * DH * 2);
    _Float16* Kh  = (_Float16*)carve((size_t)TOTAL * DH * 2);
    _Float16* Vh  = (_Float16*)carve((size_t)TOTAL * DH * 2);
    _Float16* KVt = (_Float16*)carve((size_t)NB * NH * OUT_C * OUT_C * 2);
    float* pKV    = (float*)carve((size_t)NB * NH * KSPLIT * OUT_C * OUT_C * 4);
    float* pQ     = (float*)carve(4096 * 4);
    float* pK     = (float*)carve(4096 * 4);
    float* pcs    = (float*)carve((size_t)NB * 32 * DH * 4);
    float* ksum   = (float*)carve((size_t)NB * DH * 4);
    float* scal   = (float*)carve(256);
    float* den    = (float*)carve((size_t)TOTAL * NH * 4);

    wt_kernel<<<3 * DH * IN_C / 256, 256, 0, stream>>>(Wq, Wk, Wv, Wt);
    qkv_gemm<<<dim3(DH / NT, TOTAL / MT, 3), 256, 0, stream>>>(
        x, Wt, bq, bk, bv, Qh, Kh, Vh, pQ, pK, pcs);
    reduce_norms<<<1, 256, 0, stream>>>(pQ, pK, scal);
    ksum_kernel<<<NB * DH / 256, 256, 0, stream>>>(pcs, ksum);
    den_kernel<<<TOTAL * NH / 256, 256, 0, stream>>>(Qh, ksum, scal, den);
    kv_gemm<<<dim3(NB * NH, KSPLIT), 256, 0, stream>>>(Kh, Vh, pKV);
    kv_reduce<<<NB * NH * OUT_C * OUT_C / 256, 256, 0, stream>>>(pKV, scal, KVt);
    attn_kernel<<<dim3(LSEQ / MT, NB), 256, 0, stream>>>(Qh, Vh, KVt, den, scal, out);
}
